// GatedGraph_17205638988468
// MI455X (gfx1250) — compile-verified, matrix operand order checked by reading, not run on hardware
//
#include <hip/hip_runtime.h>

// Problem constants (fixed by reference setup_inputs)
#define N_  8
#define S_  32
#define E_  6
#define L_  256
#define C_  64
#define T_  (S_*L_)       // 8192
#define D3_ (3*C_)        // 192
#define EC_ (E_*C_)       // 384 = K of the projection GEMM
#define ITER_ 10          // reference passes iteration=10 (compile-time constant)

typedef __bf16 bf16_t;
typedef bf16_t v8bf  __attribute__((ext_vector_type(8)));
typedef bf16_t v16bf __attribute__((ext_vector_type(16)));
typedef float  v8f   __attribute__((ext_vector_type(8)));

static __device__ __forceinline__ v16bf frag_cat(v8bf lo, v8bf hi) {
  return __builtin_shufflevector(lo, hi, 0,1,2,3,4,5,6,7,8,9,10,11,12,13,14,15);
}

// Load one 16x32 bf16 fragment in the documented CDNA5 A-matrix striping:
//   lanes 0-15  : row M=lane,    K = 0..7  and 16..23
//   lanes 16-31 : row M=lane-16, K = 8..15 and 24..31
// 'base' points at (row0, k0) of a row-major [rows x ld] bf16 array.
// B fragments use the same loader against K-major (transposed) storage.
static __device__ __forceinline__ v16bf load_frag(const bf16_t* base, int ld) {
  const int lane = threadIdx.x & 31;
  const int m    = lane & 15;
  const int half = lane >> 4;
  const bf16_t* p = base + m * ld + half * 8;
  v8bf lo = *(const v8bf*)(p);
  v8bf hi = *(const v8bf*)(p + 16);
  return frag_cat(lo, hi);
}

static __device__ __forceinline__ v8f wmma_bf16(v16bf a, v16bf b, v8f c) {
  return __builtin_amdgcn_wmma_f32_16x16x32_bf16(false, a, false, b, (short)0, c,
                                                 false, false);
}

static __device__ __forceinline__ float sigmoidf_(float x) {
  return 1.0f / (1.0f + __expf(-x));
}

// ---------------------------------------------------------------------------
// One-time: edge (n,s,e,k,l) fp32 -> edgeT (b,l,k) bf16  (b = (n*S+s)*E+e)
// LDS-tiled 32x32 transpose, coalesced both sides.
// ---------------------------------------------------------------------------
__global__ void __launch_bounds__(256) k_prep_edge(const float* __restrict__ edge,
                                                   bf16_t* __restrict__ edgeT) {
  __shared__ bf16_t tile[32][33];
  const int b = blockIdx.x;
  const float* src = edge + (size_t)b * (L_*L_);
  bf16_t*      dst = edgeT + (size_t)b * (L_*L_);
  const int tid = threadIdx.x;
  for (int tr = 0; tr < 8; ++tr) {
    for (int tc = 0; tc < 8; ++tc) {
      #pragma unroll
      for (int p = 0; p < 4; ++p) {
        int flat = tid + p*256;
        int r = flat >> 5, c = flat & 31;
        tile[r][c] = (bf16_t)src[(size_t)(tr*32 + r)*L_ + tc*32 + c];
      }
      __syncthreads();
      #pragma unroll
      for (int p = 0; p < 4; ++p) {
        int flat = tid + p*256;
        int r = flat >> 5, c = flat & 31;
        dst[(size_t)(tc*32 + r)*L_ + tr*32 + c] = tile[c][r];
      }
      __syncthreads();
    }
  }
}

// ---------------------------------------------------------------------------
// One-time: small weights -> bf16 transposed (K-major), plus padded-row mask.
//  WT   [192][384] : WT[d][k]    = W[k][d]   (W viewed as (384,192))
//  uzurT[128][64]  : = uz_ur[k][d]
//  uhT  [64][64]   : = uh[k][d]
//  msk  [65536]    : row all-zero in initial hidden
// ---------------------------------------------------------------------------
__global__ void __launch_bounds__(256) k_prep_small(
    const float* __restrict__ hid0, const float* __restrict__ W,
    const float* __restrict__ uzur, const float* __restrict__ uh,
    bf16_t* __restrict__ WT, bf16_t* __restrict__ uzurT,
    bf16_t* __restrict__ uhT, unsigned char* __restrict__ msk) {
  const int gid = blockIdx.x * 256 + threadIdx.x;
  if (gid < 73728) {                         // 192*384
    int d = gid / EC_, k = gid % EC_;
    WT[gid] = (bf16_t)W[(size_t)k * D3_ + d];
  } else if (gid < 81920) {                  // +128*64
    int i = gid - 73728; int d = i >> 6, k = i & 63;
    uzurT[i] = (bf16_t)uzur[(size_t)k * 128 + d];
  } else if (gid < 86016) {                  // +64*64
    int i = gid - 81920; int d = i >> 6, k = i & 63;
    uhT[i] = (bf16_t)uh[(size_t)k * 64 + d];
  } else if (gid < 151552) {                 // +65536 rows
    int r = gid - 86016;
    const float* hr = hid0 + (size_t)r * C_;
    unsigned char z = 1;
    for (int c = 0; c < C_; ++c) if (hr[c] != 0.0f) { z = 0; break; }
    msk[r] = z;
  }
}

// ---------------------------------------------------------------------------
// Per iteration: hidden fp32 -> hb (row-major bf16, A-fragment friendly)
//                            -> hbT (n,c,t bf16, B-fragment friendly)
// 64x64 LDS-tiled transpose per (n, t-tile).
// ---------------------------------------------------------------------------
__global__ void __launch_bounds__(256) k_h2bf16(const float* __restrict__ hid,
                                                bf16_t* __restrict__ hb,
                                                bf16_t* __restrict__ hbT) {
  __shared__ __align__(16) bf16_t lt[64 * 72];   // [c][t_local], stride 72
  const int n  = blockIdx.x >> 7;
  const int t0 = (blockIdx.x & 127) * 64;
  const int tid = threadIdx.x;
  #pragma unroll
  for (int it = 0; it < 4; ++it) {
    int flat4 = tid + it*256;                // 1024 float4's
    int row = flat4 >> 4;                    // t-local
    int c4  = (flat4 & 15) * 4;
    const float4 v = *(const float4*)(hid + ((size_t)(n*T_ + t0 + row))*C_ + c4);
    bf16_t b0=(bf16_t)v.x, b1=(bf16_t)v.y, b2=(bf16_t)v.z, b3=(bf16_t)v.w;
    bf16_t* ho = hb + ((size_t)(n*T_ + t0 + row))*C_ + c4;
    ho[0]=b0; ho[1]=b1; ho[2]=b2; ho[3]=b3;
    lt[(c4+0)*72 + row]=b0; lt[(c4+1)*72 + row]=b1;
    lt[(c4+2)*72 + row]=b2; lt[(c4+3)*72 + row]=b3;
  }
  __syncthreads();
  #pragma unroll
  for (int it = 0; it < 2; ++it) {
    int chunk = tid + it*256;                // 512 chunks of 8
    int c = chunk >> 3, off = (chunk & 7) * 8;
    v8bf v = *(const v8bf*)(&lt[c*72 + off]);
    *(v8bf*)(hbT + ((size_t)(n*C_ + c))*T_ + t0 + off) = v;
  }
}

// ---------------------------------------------------------------------------
// act[n,t,e*C+c] = sum_k edgeT[b][l][k] * hs[k][c] + ba[e][c]   (bf16 out)
// One block per (n,s,e): M=256, N=64, K=256. 8 waves, each 2x4 16x16 C-tiles.
// hs^T staged in LDS (64 x 256, pad 264); edgeT A-fragments streamed global.
// ---------------------------------------------------------------------------
__global__ void __launch_bounds__(256) k_msg(const bf16_t* __restrict__ edgeT,
                                             const bf16_t* __restrict__ hbT,
                                             const float* __restrict__ ba,
                                             bf16_t* __restrict__ act) {
  __shared__ __align__(16) bf16_t bs[64 * 264];   // hs^T: [c][k]
  const int b = blockIdx.x;
  const int e = b % E_;
  const int s = (b / E_) % S_;
  const int n = b / (E_ * S_);
  const int tid = threadIdx.x;
  {
    const bf16_t* src = hbT + ((size_t)n * C_) * T_ + s * L_;  // row c, stride T_
    #pragma unroll
    for (int it = 0; it < 8; ++it) {
      int chunk = tid + it*256;              // 2048 chunks of 8 bf16
      int c = chunk >> 5, off = (chunk & 31) * 8;
      *(v8bf*)(&bs[c*264 + off]) = *(const v8bf*)(src + (size_t)c*T_ + off);
    }
  }
  __syncthreads();

  const int w = tid >> 5, lane = tid & 31;
  const int mbase = w * 32;
  const bf16_t* A = edgeT + (size_t)b * (L_*L_);
  v8f acc[2][4] = {};
  for (int kt = 0; kt < 8; ++kt) {
    const int k0 = kt * 32;
    if (kt < 7) __builtin_prefetch(A + (size_t)mbase*L_ + k0 + 32, 0, 1);
    v16bf a0 = load_frag(A + (size_t)(mbase +  0)*L_ + k0, L_);
    v16bf a1 = load_frag(A + (size_t)(mbase + 16)*L_ + k0, L_);
    #pragma unroll
    for (int nt = 0; nt < 4; ++nt) {
      v16bf bfr = load_frag(&bs[(nt*16)*264 + k0], 264);
      acc[0][nt] = wmma_bf16(a0, bfr, acc[0][nt]);
      acc[1][nt] = wmma_bf16(a1, bfr, acc[1][nt]);
    }
  }
  // C/D layout: lanes 0-15 -> N=lane, M=v; lanes 16-31 -> N=lane-16, M=v+8
  const int ncol = lane & 15, half = lane >> 4;
  #pragma unroll
  for (int mt = 0; mt < 2; ++mt) {
    #pragma unroll
    for (int nt = 0; nt < 4; ++nt) {
      const int c = nt*16 + ncol;
      const float bias = ba[e*C_ + c];
      #pragma unroll
      for (int v = 0; v < 8; ++v) {
        const int t = s*L_ + mbase + mt*16 + v + 8*half;
        act[((size_t)(n*T_ + t))*EC_ + e*C_ + c] = (bf16_t)(acc[mt][nt][v] + bias);
      }
    }
  }
}

// ---------------------------------------------------------------------------
// wzrh[nt][d] = sum_K act[nt][K] * W[K][d]    M=65536 N=192 K=384 (fp32 out)
// Block: 128 rows x 192 cols; 8 waves as 4(M)x2(N); wave = 2x6 C-tiles.
// W^T K-slab (192 x 32, pad 40) staged in LDS per K-step.
// ---------------------------------------------------------------------------
__global__ void __launch_bounds__(256) k_proj(const bf16_t* __restrict__ act,
                                              const bf16_t* __restrict__ WT,
                                              float* __restrict__ wzrh) {
  __shared__ __align__(16) bf16_t bs[192 * 40];
  const int row0 = blockIdx.x * 128;
  const int tid = threadIdx.x;
  const int w = tid >> 5, lane = tid & 31;
  const int mbase = row0 + (w >> 1) * 32;
  const int nbase = (w & 1) * 96;
  v8f acc[2][6] = {};
  for (int kt = 0; kt < 12; ++kt) {
    const int k0 = kt * 32;
    __syncthreads();
    #pragma unroll
    for (int it = 0; it < 3; ++it) {
      int chunk = tid + it*256;              // 768 chunks (192 rows x 4)
      int d = chunk >> 2, off = (chunk & 3) * 8;
      *(v8bf*)(&bs[d*40 + off]) = *(const v8bf*)(WT + (size_t)d*EC_ + k0 + off);
    }
    __syncthreads();
    v16bf a0 = load_frag(act + (size_t)(mbase +  0)*EC_ + k0, EC_);
    v16bf a1 = load_frag(act + (size_t)(mbase + 16)*EC_ + k0, EC_);
    #pragma unroll
    for (int nt = 0; nt < 6; ++nt) {
      v16bf bfr = load_frag(&bs[(nbase + nt*16)*40], 40);
      acc[0][nt] = wmma_bf16(a0, bfr, acc[0][nt]);
      acc[1][nt] = wmma_bf16(a1, bfr, acc[1][nt]);
    }
  }
  const int ncol = lane & 15, half = lane >> 4;
  #pragma unroll
  for (int mt = 0; mt < 2; ++mt)
    #pragma unroll
    for (int nt = 0; nt < 6; ++nt)
      #pragma unroll
      for (int v = 0; v < 8; ++v) {
        const int r = mbase + mt*16 + v + 8*half;
        const int d = nbase + nt*16 + ncol;
        wzrh[(size_t)r*D3_ + d] = acc[mt][nt][v];
      }
}

// ---------------------------------------------------------------------------
// Fused GRU gate update (in place on hidden fp32):
//   uzur = h @ uz_ur (WMMA) ; z=sig(wz+uz), r=sig(wr+ur)
//   rh -> LDS (bf16, A-fragment layout) ; htilde = tanh(wh + rh @ uh (WMMA))
//   h = (1-z)h + z*htilde ; padded rows zeroed.
// Block = 128 rows, 8 waves x 16 rows.
// ---------------------------------------------------------------------------
__global__ void __launch_bounds__(256) k_gate(const float* __restrict__ wzrh,
                                              const bf16_t* __restrict__ hb,
                                              const bf16_t* __restrict__ uzurT,
                                              const bf16_t* __restrict__ uhT,
                                              const unsigned char* __restrict__ msk,
                                              float* __restrict__ hid) {
  __shared__ __align__(16) bf16_t su[128 * 72];   // uz_ur^T [d][k]
  __shared__ __align__(16) bf16_t sh[64 * 72];    // uh^T    [d][k]
  __shared__ __align__(16) bf16_t srh[128 * 72];  // r*h     [m][c]
  const int row0 = blockIdx.x * 128;
  const int tid = threadIdx.x;
  #pragma unroll
  for (int it = 0; it < 4; ++it) {
    int chunk = tid + it*256;                // 1024 chunks
    int d = chunk >> 3, off = (chunk & 7) * 8;
    *(v8bf*)(&su[d*72 + off]) = *(const v8bf*)(uzurT + d*C_ + off);
  }
  #pragma unroll
  for (int it = 0; it < 2; ++it) {
    int chunk = tid + it*256;                // 512 chunks
    int d = chunk >> 3, off = (chunk & 7) * 8;
    *(v8bf*)(&sh[d*72 + off]) = *(const v8bf*)(uhT + d*C_ + off);
  }
  __syncthreads();

  const int w = tid >> 5, lane = tid & 31;
  const int r0 = row0 + w * 16;
  const int ncol = lane & 15, half = lane >> 4;

  // GEMM1: (16x64) @ (64x128)
  v8f acc1[8] = {};
  #pragma unroll
  for (int kt = 0; kt < 2; ++kt) {
    const int k0 = kt * 32;
    v16bf a = load_frag(hb + (size_t)r0*C_ + k0, C_);
    #pragma unroll
    for (int nt = 0; nt < 8; ++nt) {
      v16bf bfr = load_frag(&su[(nt*16)*72 + k0], 72);
      acc1[nt] = wmma_bf16(a, bfr, acc1[nt]);
    }
  }

  float zr[4][8];
  #pragma unroll
  for (int j = 0; j < 4; ++j) {
    const int c = j*16 + ncol;               // uz: tiles 0-3, ur: tiles 4-7
    #pragma unroll
    for (int v = 0; v < 8; ++v) {
      const int r = r0 + v + 8*half;
      const float wz = wzrh[(size_t)r*D3_ + c];
      const float wr = wzrh[(size_t)r*D3_ + C_ + c];
      const float z  = sigmoidf_(wz + acc1[j][v]);
      const float rr = sigmoidf_(wr + acc1[j+4][v]);
      zr[j][v] = z;
      const float h = hid[(size_t)r*C_ + c];
      srh[(w*16 + v + 8*half)*72 + c] = (bf16_t)(rr * h);
    }
  }
  __syncthreads();

  // GEMM2: (16x64) @ (64x64)
  v8f acc2[4] = {};
  #pragma unroll
  for (int kt = 0; kt < 2; ++kt) {
    const int k0 = kt * 32;
    v16bf a = load_frag(&srh[(w*16)*72 + k0], 72);
    #pragma unroll
    for (int nt = 0; nt < 4; ++nt) {
      v16bf bfr = load_frag(&sh[(nt*16)*72 + k0], 72);
      acc2[nt] = wmma_bf16(a, bfr, acc2[nt]);
    }
  }

  #pragma unroll
  for (int j = 0; j < 4; ++j) {
    const int c = j*16 + ncol;
    #pragma unroll
    for (int v = 0; v < 8; ++v) {
      const int r = r0 + v + 8*half;
      const float wh = wzrh[(size_t)r*D3_ + 2*C_ + c];
      const float ht = tanhf(wh + acc2[j][v]);
      const float h  = hid[(size_t)r*C_ + c];
      const float z  = zr[j][v];
      float hn = (1.0f - z)*h + z*ht;
      if (msk[r]) hn = 0.0f;
      hid[(size_t)r*C_ + c] = hn;
    }
  }
}

// ---------------------------------------------------------------------------
extern "C" void kernel_launch(void* const* d_in, const int* in_sizes, int n_in,
                              void* d_out, int out_size, void* d_ws, size_t ws_size,
                              hipStream_t stream) {
  (void)in_sizes; (void)n_in; (void)out_size; (void)ws_size;
  const float* hid0 = (const float*)d_in[0];
  const float* edge = (const float*)d_in[1];
  const float* ba   = (const float*)d_in[2];
  const float* W    = (const float*)d_in[3];
  const float* uzur = (const float*)d_in[4];
  const float* uh   = (const float*)d_in[5];
  float* hid = (float*)d_out;   // hidden state lives in d_out (updated in place)

  // Workspace carve-up (256B aligned). Total ~319 MB.
  char* ws = (char*)d_ws;
  size_t off = 0;
  auto carve = [&](size_t bytes) {
    void* p = ws + off;
    off += (bytes + 255) & ~(size_t)255;
    return p;
  };
  bf16_t* edgeT   = (bf16_t*)carve((size_t)N_*S_*E_*L_*L_ * 2);  // 201.3 MB
  bf16_t* hb      = (bf16_t*)carve((size_t)N_*T_*C_ * 2);        // 8.4 MB
  bf16_t* hbT     = (bf16_t*)carve((size_t)N_*T_*C_ * 2);        // 8.4 MB
  bf16_t* actb    = (bf16_t*)carve((size_t)N_*T_*EC_ * 2);       // 50.3 MB
  float*  wzrh    = (float*) carve((size_t)N_*T_*D3_ * 4);       // 50.3 MB
  bf16_t* WT      = (bf16_t*)carve((size_t)EC_*D3_ * 2);
  bf16_t* uzurT   = (bf16_t*)carve((size_t)128*64 * 2);
  bf16_t* uhT     = (bf16_t*)carve((size_t)64*64 * 2);
  unsigned char* msk = (unsigned char*)carve((size_t)N_*T_);

  // hidden state := input hidden (d2d copy is graph-capture safe)
  hipMemcpyAsync(hid, hid0, (size_t)N_*T_*C_ * sizeof(float),
                 hipMemcpyDeviceToDevice, stream);

  // One-time preprocessing (amortized over 10 iterations)
  k_prep_edge <<<N_*S_*E_, 256, 0, stream>>>(edge, edgeT);
  k_prep_small<<<592,      256, 0, stream>>>(hid0, W, uzur, uh, WT, uzurT, uhT, msk);

  for (int it = 0; it < ITER_; ++it) {
    k_h2bf16<<<N_*128,        256, 0, stream>>>(hid, hb, hbT);
    k_msg   <<<N_*S_*E_,      256, 0, stream>>>(edgeT, hbT, ba, actb);
    k_proj  <<<(N_*T_)/128,   256, 0, stream>>>(actb, WT, wzrh);
    k_gate  <<<(N_*T_)/128,   256, 0, stream>>>(wzrh, hb, uzurT, uhT, msk, hid);
  }
}
